// SwinTransformer3DStage_52750788329881
// MI455X (gfx1250) — matrix operand order repaired and checked by reading, not compiled, not run on hardware
//
#include <hip/hip_runtime.h>
#include <hip/hip_bf16.h>
#include <math.h>

// ---------------- constants ----------------
constexpr int CC    = 128;
constexpr int HEADS = 4;
constexpr int NWIN  = 98;     // tokens per window (2*7*7)
constexpr int NPAD  = 112;    // padded to 7*16
constexpr int NWTOT = 2048;   // 8 * 4 * 8 * 8 windows
constexpr int TOK   = 200704; // 8*8*56*56
constexpr int MPAD  = NWTOT * NPAD; // 229376

// ---------------- WMMA types ----------------
typedef __attribute__((ext_vector_type(16))) __bf16 v16bf;
typedef __attribute__((ext_vector_type(8)))  float  v8f;

union Frag { unsigned u[8]; v16bf v; };

__device__ __forceinline__ unsigned short f2bf(float f) {
  unsigned u = __float_as_uint(f);
  unsigned r = u + 0x7FFFu + ((u >> 16) & 1u);
  return (unsigned short)(r >> 16);
}
__device__ __forceinline__ float bf2f(unsigned short h) {
  return __uint_as_float(((unsigned)h) << 16);
}
// A-fragment K index, contiguous order k = 16*half + 2*v (matches B fills)
__device__ __forceinline__ int kA(int v, int half) {
  return 2 * v + 16 * half;
}
// window (wi, n) -> flat token index in [B,T,H,W] (channel-last), with shift roll
__device__ __forceinline__ int wmap(int wi, int n, int shifted) {
  int b = wi >> 8;
  int r = wi & 255;
  int wt = r >> 6, wh = (r >> 3) & 7, ww = r & 7;
  int it = n / 49, r3 = n % 49, ih = r3 / 7, iw = r3 % 7;
  int tt = wt * 2 + it, hh = wh * 7 + ih, wc = ww * 7 + iw;
  if (shifted) {
    tt = (tt + 1) & 7;
    hh += 3; if (hh >= 56) hh -= 56;
    wc += 3; if (wc >= 56) wc -= 56;
  }
  return ((b * 8 + tt) * 56 + hh) * 56 + wc;
}

// ---- CDNA5 async copy: global -> LDS (16B), tracked by ASYNCcnt ----
// GVS addressing: mem = SADDR(64) + VGPR_off(i32) ; VDST VGPR = per-lane LDS byte addr.
__device__ __forceinline__ void async_ld16(unsigned lds_addr, const void* sbase, int voff) {
  asm volatile("global_load_async_to_lds_b128 %0, %1, %2"
               :
               : "v"(lds_addr), "v"(voff), "s"(sbase)
               : "memory");
}
__device__ __forceinline__ void wait_async_le4() {
  asm volatile("s_wait_asynccnt 0x4" ::: "memory");
}
__device__ __forceinline__ void wait_async_0() {
  asm volatile("s_wait_asynccnt 0x0" ::: "memory");
}
// low 32 bits of a generic pointer into the LDS aperture == LDS byte offset
__device__ __forceinline__ unsigned lds_off(const void* p) {
  return (unsigned)(size_t)p;
}

// ---------------- prep kernels ----------------
// x [B,C,T,H,W] fp32 -> xact [token][C] fp32
__global__ __launch_bounds__(256) void transpose_in_kernel(const float* __restrict__ x,
                                                           float* __restrict__ xact) {
  long i = (long)blockIdx.x * 256 + threadIdx.x;
  if (i >= (long)12845056) return;
  int w = i % 56; long r = i / 56;
  int h = r % 56; r /= 56;
  int t = r % 8;  r /= 8;
  int c = r % 128;
  int b = (int)(r / 128);
  int token = ((b * 8 + t) * 56 + h) * 56 + w;
  xact[(size_t)token * CC + c] = x[i];
}
// xact -> d_out [B,C,T,H,W]
__global__ __launch_bounds__(256) void final_kernel(const float* __restrict__ xact,
                                                    float* __restrict__ out) {
  long i = (long)blockIdx.x * 256 + threadIdx.x;
  if (i >= (long)12845056) return;
  int w = i % 56; long r = i / 56;
  int h = r % 56; r /= 56;
  int t = r % 8;  r /= 8;
  int c = r % 128;
  int b = (int)(r / 128);
  int token = ((b * 8 + t) * 56 + h) * 56 + w;
  out[i] = xact[(size_t)token * CC + c];
}
// src fp32 [K][N] -> dst bf16 [N][K]
__global__ __launch_bounds__(256) void transpose_w_kernel(unsigned short* __restrict__ dst,
                                                          const float* __restrict__ src,
                                                          int K, int N) {
  int i = blockIdx.x * 256 + threadIdx.x;
  if (i >= K * N) return;
  int n = i / K, k = i % K;
  dst[i] = f2bf(src[k * N + n]);
}
// rpb [2][507][4] -> biasTab [2][4][98][98]
__global__ __launch_bounds__(256) void buildbias_kernel(const float* __restrict__ rpb,
                                                        float* __restrict__ biasTab) {
  int i = blockIdx.x * 256 + threadIdx.x;
  if (i >= 2 * HEADS * NWIN * NWIN) return;
  int m = i % NWIN;
  int n = (i / NWIN) % NWIN;
  int h = (i / (NWIN * NWIN)) % HEADS;
  int l = i / (NWIN * NWIN * HEADS);
  int t1 = n / 49, h1 = (n / 7) % 7, w1 = n % 7;
  int t2 = m / 49, h2 = (m / 7) % 7, w2 = m % 7;
  int rel = (t1 - t2 + 1) * 169 + (h1 - h2 + 6) * 13 + (w1 - w2 + 6);
  biasTab[i] = rpb[((size_t)l * 507 + rel) * HEADS + h];
}

// ---------------- LayerNorm (one wave per row) ----------------
__global__ __launch_bounds__(256) void ln_kernel(const float* __restrict__ xact,
                                                 const float* __restrict__ g,
                                                 const float* __restrict__ b,
                                                 unsigned short* __restrict__ out,
                                                 int nrows, int windowed, int shifted) {
  int row = blockIdx.x * 8 + (threadIdx.x >> 5);
  if (row >= nrows) return;
  int lane = threadIdx.x & 31;
  unsigned* outu = (unsigned*)(out + (size_t)row * CC);
  int token;
  if (windowed) {
    int wi = row / NPAD, n = row % NPAD;
    if (n >= NWIN) { outu[lane * 2] = 0u; outu[lane * 2 + 1] = 0u; return; }
    token = wmap(wi, n, shifted);
  } else {
    token = row;
  }
  float4 xv = *(const float4*)&xact[(size_t)token * CC + lane * 4];
  float s = xv.x + xv.y + xv.z + xv.w;
  s += __shfl_xor(s, 16, 32); s += __shfl_xor(s, 8, 32);
  s += __shfl_xor(s, 4, 32);  s += __shfl_xor(s, 2, 32); s += __shfl_xor(s, 1, 32);
  float mean = s * (1.0f / 128.0f);
  float q = xv.x * xv.x + xv.y * xv.y + xv.z * xv.z + xv.w * xv.w;
  q += __shfl_xor(q, 16, 32); q += __shfl_xor(q, 8, 32);
  q += __shfl_xor(q, 4, 32);  q += __shfl_xor(q, 2, 32); q += __shfl_xor(q, 1, 32);
  float var = q * (1.0f / 128.0f) - mean * mean;
  float rstd = rsqrtf(var + 1e-5f);
  int c0 = lane * 4;
  float y0 = (xv.x - mean) * rstd * g[c0 + 0] + b[c0 + 0];
  float y1 = (xv.y - mean) * rstd * g[c0 + 1] + b[c0 + 1];
  float y2 = (xv.z - mean) * rstd * g[c0 + 2] + b[c0 + 2];
  float y3 = (xv.w - mean) * rstd * g[c0 + 3] + b[c0 + 3];
  outu[lane * 2]     = (unsigned)f2bf(y0) | ((unsigned)f2bf(y1) << 16);
  outu[lane * 2 + 1] = (unsigned)f2bf(y2) | ((unsigned)f2bf(y3) << 16);
}

// ---------------- generic WMMA GEMM: D = A[M,K] * Bt[N,K]^T + bias ----------------
// Async-to-LDS double-buffered pipeline: while tile k is consumed by 4 WMMAs,
// tile k+1 streams into the other LDS buffer via GLOBAL_LOAD_ASYNC_TO_LDS_B128.
// MODE 0: store bf16        MODE 1: GELU -> bf16
// MODE 2: xact[row] += val  MODE 3: window-reverse scatter add into xact
template <int MODE>
__global__ __launch_bounds__(128) void gemm_wmma(const unsigned short* __restrict__ A,
                                                 const unsigned short* __restrict__ Bt,
                                                 const float* __restrict__ bias,
                                                 int K,
                                                 unsigned short* __restrict__ outB, int ldout,
                                                 float* __restrict__ xact, int shifted) {
  __shared__ alignas(16) unsigned short As[2][64][32];
  __shared__ alignas(16) unsigned short Bs[2][64][32];
  int tid = threadIdx.x, wave = tid >> 5, lane = tid & 31;
  int half = lane >> 4, l15 = lane & 31 & 15;
  l15 = lane & 15;
  int rBase = blockIdx.x * 64, cBase = blockIdx.y * 64;

  // staging coords: each thread moves 2 x 16B per matrix per K-tile
  int r = tid >> 1;            // row within 64-row tile
  int qa = (tid & 1) * 2;      // which pair of 16B chunks in the 64B row
  unsigned ldsA[2], ldsB[2];
  ldsA[0] = lds_off(&As[0][r][qa * 8]);
  ldsA[1] = lds_off(&As[1][r][qa * 8]);
  ldsB[0] = lds_off(&Bs[0][r][qa * 8]);
  ldsB[1] = lds_off(&Bs[1][r][qa * 8]);
  int offA0 = (int)((((size_t)(rBase + r)) * K + qa * 8) * 2); // byte offsets (k0 added later)
  int offB0 = (int)((((size_t)(cBase + r)) * K + qa * 8) * 2);

  v8f acc[4];
  v8f zero = {0.f, 0.f, 0.f, 0.f, 0.f, 0.f, 0.f, 0.f};
#pragma unroll
  for (int nt = 0; nt < 4; ++nt) acc[nt] = zero;

  int nT = K >> 5;
  // prologue: stream tile 0 into buffer 0 (4 async ops -> ASYNCcnt = 4)
  async_ld16(ldsA[0], A, offA0);
  async_ld16(ldsA[0] + 16, A, offA0 + 16);
  async_ld16(ldsB[0], Bt, offB0);
  async_ld16(ldsB[0] + 16, Bt, offB0 + 16);

  for (int it = 0; it < nT; ++it) {
    int cur = it & 1;
    if (it + 1 < nT) {
      int kb = (it + 1) * 64; // byte offset of next K-tile (32 elems * 2B)
      async_ld16(ldsA[cur ^ 1], A, offA0 + kb);
      async_ld16(ldsA[cur ^ 1] + 16, A, offA0 + kb + 16);
      async_ld16(ldsB[cur ^ 1], Bt, offB0 + kb);
      async_ld16(ldsB[cur ^ 1] + 16, Bt, offB0 + kb + 16);
      wait_async_le4(); // retire current tile's 4 loads (in-order completion)
    } else {
      wait_async_0();
    }
    __syncthreads(); // all waves' current tile resident in LDS

    Frag a;
#pragma unroll
    for (int v = 0; v < 8; ++v)
      a.u[v] = *(const unsigned*)&As[cur][wave * 16 + l15][kA(v, half)];
#pragma unroll
    for (int nt = 0; nt < 4; ++nt) {
      Frag bb;
#pragma unroll
      for (int v = 0; v < 8; ++v)
        bb.u[v] = *(const unsigned*)&Bs[cur][nt * 16 + l15][2 * v + 16 * half];
      acc[nt] = __builtin_amdgcn_wmma_f32_16x16x32_bf16(false, a.v, false, bb.v,
                                                        (short)0, acc[nt], false, false);
    }
    __syncthreads(); // done reading 'cur'; next iter may stream into it
  }

#pragma unroll
  for (int nt = 0; nt < 4; ++nt) {
    int col = cBase + nt * 16 + l15;
    float bs = bias[col];
#pragma unroll
    for (int j = 0; j < 8; ++j) {
      int row = rBase + wave * 16 + j + 8 * half;
      float val = acc[nt][j] + bs;
      if (MODE == 0) {
        outB[(size_t)row * ldout + col] = f2bf(val);
      } else if (MODE == 1) {
        float ge = 0.5f * val * (1.0f + erff(val * 0.70710678118654752f));
        outB[(size_t)row * ldout + col] = f2bf(ge);
      } else if (MODE == 2) {
        xact[(size_t)row * CC + col] += val;
      } else { // MODE 3
        int wi = row / NPAD, n = row % NPAD;
        if (n < NWIN) {
          int tok = wmap(wi, n, shifted);
          xact[(size_t)tok * CC + col] += val;
        }
      }
    }
  }
}

// ---------------- windowed attention: one block per (window, head) ----------------
__global__ __launch_bounds__(224) void attn_kernel(const unsigned short* __restrict__ qkv,
                                                   const float* __restrict__ biasT,
                                                   unsigned short* __restrict__ out,
                                                   int shifted) {
  int wi = blockIdx.x, h = blockIdx.y;
  int tid = threadIdx.x, wave = tid >> 5, lane = tid & 31;
  int half = lane >> 4, l15 = lane & 15;

  __shared__ unsigned short Vt[32][128];       // V^T, m padded to 128 (zeros)
  __shared__ unsigned short Pst[7][16][128];   // per-wave softmax(P) strips (bf16)
  __shared__ int labels[NPAD];

  // zero V^T then fill transposed from qkv (v block at col 256 + h*32)
  for (int i = tid; i < 32 * 128 / 2; i += 224) ((unsigned*)Vt)[i] = 0u;
  for (int i = tid; i < NWIN * 16; i += 224) {
    int m = i >> 4, dp = i & 15;
    unsigned val = *(const unsigned*)&qkv[(size_t)(wi * NPAD + m) * 384 + 256 + h * 32 + dp * 2];
    Vt[dp * 2][m]     = (unsigned short)(val & 0xffffu);
    Vt[dp * 2 + 1][m] = (unsigned short)(val >> 16);
  }
  if (shifted && tid < NPAD) { // shift-mask region labels in the rolled frame
    int n = tid;
    int r = wi & 255;
    int wt = r >> 6, wh = (r >> 3) & 7, ww = r & 7;
    int it = n / 49, r3 = n % 49, ih = r3 / 7, iw = r3 % 7;
    int tt = wt * 2 + it, hh = wh * 7 + ih, wc = ww * 7 + iw;
    int st = tt < 6 ? 0 : (tt < 7 ? 1 : 2);
    int sh = hh < 49 ? 0 : (hh < 53 ? 1 : 2);
    int sw = wc < 49 ? 0 : (wc < 53 ? 1 : 2);
    labels[n] = (st * 3 + sh) * 3 + sw;
  }
  __syncthreads();

  // ---- S = Q K^T (head_dim 32 = one WMMA K step), Q/K fragments from global ----
  Frag aQ;
  int qrow = wi * NPAD + wave * 16 + l15;
#pragma unroll
  for (int v = 0; v < 8; ++v)
    aQ.u[v] = *(const unsigned*)&qkv[(size_t)qrow * 384 + h * 32 + kA(v, half)];

  v8f zero = {0.f, 0.f, 0.f, 0.f, 0.f, 0.f, 0.f, 0.f};
  v8f accS[7];
#pragma unroll
  for (int t = 0; t < 7; ++t) {
    Frag bK;
    int krow = wi * NPAD + t * 16 + l15;
#pragma unroll
    for (int v = 0; v < 8; ++v)
      bK.u[v] = *(const unsigned*)&qkv[(size_t)krow * 384 + 128 + h * 32 + 2 * v + 16 * half];
    accS[t] = __builtin_amdgcn_wmma_f32_16x16x32_bf16(false, aQ.v, false, bK.v,
                                                      (short)0, zero, false, false);
  }

  // ---- scale + rel-pos bias + shift mask ----
  const float scale = 0.17677669529663687f; // 32^-0.5
#pragma unroll
  for (int t = 0; t < 7; ++t) {
    int m = t * 16 + l15;
#pragma unroll
    for (int j = 0; j < 8; ++j) {
      int n = wave * 16 + j + 8 * half;
      float s = accS[t][j] * scale;
      if (m < NWIN) {
        int nn = n < NWIN ? n : 0;
        s += biasT[(h * NWIN + nn) * NWIN + m];
        if (shifted && (labels[nn] != labels[m])) s += -100.0f;
      } else {
        s = -1e30f; // padded key columns
      }
      accS[t][j] = s;
    }
  }

  // ---- softmax per row (rows live in half-wave lane groups) ----
#pragma unroll
  for (int j = 0; j < 8; ++j) {
    float mx = -1e30f;
#pragma unroll
    for (int t = 0; t < 7; ++t) mx = fmaxf(mx, accS[t][j]);
    mx = fmaxf(mx, __shfl_xor(mx, 1, 16));
    mx = fmaxf(mx, __shfl_xor(mx, 2, 16));
    mx = fmaxf(mx, __shfl_xor(mx, 4, 16));
    mx = fmaxf(mx, __shfl_xor(mx, 8, 16));
    float sum = 0.f;
#pragma unroll
    for (int t = 0; t < 7; ++t) {
      float p = __expf(accS[t][j] - mx);
      accS[t][j] = p;
      sum += p;
    }
    sum += __shfl_xor(sum, 1, 16);
    sum += __shfl_xor(sum, 2, 16);
    sum += __shfl_xor(sum, 4, 16);
    sum += __shfl_xor(sum, 8, 16);
    float inv = 1.0f / sum;
    int rr = j + 8 * half;
#pragma unroll
    for (int t = 0; t < 7; ++t)
      Pst[wave][rr][t * 16 + l15] = f2bf(accS[t][j] * inv);
  }
  // zero padded P columns 112..127
  for (int i = lane; i < 16 * 8; i += 32) {
    int r = i >> 3, cq = i & 7;
    *(unsigned*)&Pst[wave][r][112 + cq * 2] = 0u;
  }
  __syncthreads();

  // ---- O = P V  (K = 128 padded tokens, 4 WMMA K steps, 2 N tiles) ----
  v8f accO[2];
  accO[0] = zero; accO[1] = zero;
#pragma unroll
  for (int s = 0; s < 4; ++s) {
    Frag aP;
#pragma unroll
    for (int v = 0; v < 8; ++v)
      aP.u[v] = *(const unsigned*)&Pst[wave][l15][s * 32 + kA(v, half)];
#pragma unroll
    for (int nt = 0; nt < 2; ++nt) {
      Frag bV;
#pragma unroll
      for (int v = 0; v < 8; ++v)
        bV.u[v] = *(const unsigned*)&Vt[nt * 16 + l15][s * 32 + 2 * v + 16 * half];
      accO[nt] = __builtin_amdgcn_wmma_f32_16x16x32_bf16(false, aP.v, false, bV.v,
                                                         (short)0, accO[nt], false, false);
    }
  }
#pragma unroll
  for (int nt = 0; nt < 2; ++nt) {
#pragma unroll
    for (int j = 0; j < 8; ++j) {
      int n = wave * 16 + j + 8 * half;
      if (n < NWIN)
        out[(size_t)(wi * NPAD + n) * CC + h * 32 + nt * 16 + l15] = f2bf(accO[nt][j]);
    }
  }
}

// ---------------- host orchestration ----------------
extern "C" void kernel_launch(void* const* d_in, const int* in_sizes, int n_in,
                              void* d_out, int out_size, void* d_ws, size_t ws_size,
                              hipStream_t stream) {
  (void)in_sizes; (void)n_in; (void)out_size; (void)ws_size;
  const float* x      = (const float*)d_in[0];
  const float* ln1_g  = (const float*)d_in[1];
  const float* ln1_b  = (const float*)d_in[2];
  const float* qkv_w  = (const float*)d_in[3];
  const float* qkv_b  = (const float*)d_in[4];
  const float* rpb    = (const float*)d_in[5];
  const float* proj_w = (const float*)d_in[6];
  const float* proj_b = (const float*)d_in[7];
  const float* ln2_g  = (const float*)d_in[8];
  const float* ln2_b  = (const float*)d_in[9];
  const float* fc1_w  = (const float*)d_in[10];
  const float* fc1_b  = (const float*)d_in[11];
  const float* fc2_w  = (const float*)d_in[12];
  const float* fc2_b  = (const float*)d_in[13];

  char* ws = (char*)d_ws;
  float*          xact    = (float*)(ws);                        // 200704*128*4
  unsigned short* buf1    = (unsigned short*)(ws + 102760448UL); // xw / attn_out / xln2
  unsigned short* buf2    = (unsigned short*)(ws + 161480704UL); // qkv / mlp hidden
  unsigned short* wT      = (unsigned short*)(ws + 367001600UL); // bf16 B^T weights
  float*          biasTab = (float*)(ws + 367788032UL);          // [2][4][98][98]

  // prep: residual layout + weights + rel-pos bias tables
  transpose_in_kernel<<<(12845056 + 255) / 256, 256, 0, stream>>>(x, xact);
  buildbias_kernel<<<(2 * HEADS * NWIN * NWIN + 255) / 256, 256, 0, stream>>>(rpb, biasTab);
  for (int l = 0; l < 2; ++l) {
    unsigned short* qkvT = wT + (size_t)l * 196608;
    unsigned short* projT = qkvT + 49152;
    unsigned short* fc1T  = qkvT + 65536;
    unsigned short* fc2T  = qkvT + 131072;
    transpose_w_kernel<<<(49152 + 255) / 256, 256, 0, stream>>>(qkvT, qkv_w + (size_t)l * 49152, 128, 384);
    transpose_w_kernel<<<(16384 + 255) / 256, 256, 0, stream>>>(projT, proj_w + (size_t)l * 16384, 128, 128);
    transpose_w_kernel<<<(65536 + 255) / 256, 256, 0, stream>>>(fc1T, fc1_w + (size_t)l * 65536, 128, 512);
    transpose_w_kernel<<<(65536 + 255) / 256, 256, 0, stream>>>(fc2T, fc2_w + (size_t)l * 65536, 512, 128);
  }

  for (int l = 0; l < 2; ++l) {
    int shifted = l & 1;
    unsigned short* qkvT = wT + (size_t)l * 196608;
    unsigned short* projT = qkvT + 49152;
    unsigned short* fc1T  = qkvT + 65536;
    unsigned short* fc2T  = qkvT + 131072;

    // LN1 + roll + window partition -> bf16 (padded windows, zero pad rows)
    ln_kernel<<<MPAD / 8, 256, 0, stream>>>(xact, ln1_g + l * 128, ln1_b + l * 128,
                                            buf1, MPAD, 1, shifted);
    // QKV GEMM: [229376,128] x [128,384]
    gemm_wmma<0><<<dim3(MPAD / 64, 6), 128, 0, stream>>>(buf1, qkvT, qkv_b + l * 384,
                                                         128, buf2, 384, nullptr, 0);
    // windowed attention
    attn_kernel<<<dim3(NWTOT, HEADS), 224, 0, stream>>>(buf2,
                                                        biasTab + (size_t)l * HEADS * NWIN * NWIN,
                                                        buf1, shifted);
    // proj GEMM + window reverse + un-roll + residual add
    gemm_wmma<3><<<dim3(MPAD / 64, 2), 128, 0, stream>>>(buf1, projT, proj_b + l * 128,
                                                         128, nullptr, 0, xact, shifted);
    // LN2 -> bf16
    ln_kernel<<<TOK / 8, 256, 0, stream>>>(xact, ln2_g + l * 128, ln2_b + l * 128,
                                           buf1, TOK, 0, 0);
    // fc1 GEMM + exact GELU
    gemm_wmma<1><<<dim3(TOK / 64, 8), 128, 0, stream>>>(buf1, fc1T, fc1_b + l * 512,
                                                        128, buf2, 512, nullptr, 0);
    // fc2 GEMM + residual add
    gemm_wmma<2><<<dim3(TOK / 64, 2), 128, 0, stream>>>(buf2, fc2T, fc2_b + l * 128,
                                                        512, nullptr, 0, xact, 0);
  }
  final_kernel<<<(12845056 + 255) / 256, 256, 0, stream>>>(xact, (float*)d_out);
}
